// GRU_6717328851359
// MI455X (gfx1250) — matrix operand order repaired and checked by reading, not compiled, not run on hardware
//
#include <hip/hip_runtime.h>
#include <hip/hip_bf16.h>

#define T_DIM 4096
#define D_DIM 1024
#define NG    3072   // 3 * D

typedef __attribute__((ext_vector_type(16))) __bf16 v16bf;
typedef __attribute__((ext_vector_type(8)))  float  v8f;

static __device__ __forceinline__ unsigned short f2bf(float f) {
  unsigned int u = __float_as_uint(f);
  u += 0x7FFFu + ((u >> 16) & 1u);          // round-to-nearest-even
  return (unsigned short)(u >> 16);
}
static __device__ __forceinline__ float bf2f(unsigned short s) {
  return __uint_as_float(((unsigned int)s) << 16);
}

// ---------------- elementwise converts ----------------
__global__ void k_f32_to_bf16(const float* __restrict__ in,
                              unsigned short* __restrict__ out, int n) {
  int i = blockIdx.x * blockDim.x + threadIdx.x;
  if (i < n) out[i] = f2bf(in[i]);
}

__global__ void k_relu_to_bf16(const float* __restrict__ in,
                               unsigned short* __restrict__ out, int n) {
  int i = blockIdx.x * blockDim.x + threadIdx.x;
  if (i < n) out[i] = f2bf(fmaxf(in[i], 0.0f));
}

// ---------------- weight transpose+convert: WT[g*D + j][k] = bf16(Wg[k][j]) ----
__global__ void k_transpose_bf16(const float* __restrict__ W0,
                                 const float* __restrict__ W1,
                                 const float* __restrict__ W2,
                                 unsigned short* __restrict__ WT) {
  __shared__ float tile[32][33];
  const int g = blockIdx.z;
  const float* W = (g == 0) ? W0 : (g == 1) ? W1 : W2;
  const int j0 = blockIdx.x * 32, k0 = blockIdx.y * 32;
  const int tx = threadIdx.x & 31, ty = threadIdx.x >> 5;   // 1024 threads
  tile[ty][tx] = W[(size_t)(k0 + ty) * D_DIM + j0 + tx];    // coalesced read
  __syncthreads();
  WT[((size_t)g * D_DIM + j0 + ty) * D_DIM + k0 + tx] = f2bf(tile[tx][ty]);
}

// ---------------- init: barrier, h0, packed input-gate bias ----------------
__global__ void k_init(const float* __restrict__ init_state,
                       const float* __restrict__ bir,
                       const float* __restrict__ biz,
                       const float* __restrict__ bin_,
                       float* __restrict__ hA, float* __restrict__ biasg,
                       int* __restrict__ bar) {
  int i = blockIdx.x * blockDim.x + threadIdx.x;
  if (i == 0) *bar = 0;
  if (i < D_DIM) {
    hA[i] = init_state[i];
    biasg[i]            = bir[i];
    biasg[D_DIM + i]    = biz[i];
    biasg[2 * D_DIM + i] = bin_[i];
  }
}

// ---------------- bf16 WMMA GEMM: C[M,N] = A[M,K] * BT[N,K]^T + bias ----------
// one wave computes a 16x64 tile (4 WMMA accumulators, A reused 4x)
__global__ void __launch_bounds__(256)
k_wmma_gemm(const unsigned short* __restrict__ A,
            const unsigned short* __restrict__ BT,
            const float* __restrict__ bias,
            float* __restrict__ C, int M, int N, int K) {
  const int lane  = threadIdx.x & 31;
  const int wave  = (blockIdx.x * blockDim.x + threadIdx.x) >> 5;
  const int tilesN = N >> 6;
  const int jobs   = (M >> 4) * tilesN;
  if (wave >= jobs) return;                       // wave-uniform: EXEC stays all-1
  const int m0 = (wave / tilesN) << 4;
  const int n0 = (wave % tilesN) << 6;

  // A 16x32 bf16, contiguous k order chosen here (not the ISA table order): lanes 0-15 -> K 0..15; lanes 16-31 -> K 16..31
  const int mRow = m0 + (lane & 15);
  const int kA   = (lane >> 4) << 4;              // 0 or 16
  // B 32x16 bf16, same k order as A: lanes 0-15 -> K 0..15; lanes 16-31 -> K 16..31 (contiguous in BT row)
  const int nCol = lane & 15;
  const int kB   = (lane >> 4) << 4;              // 0 or 16

  const uint4* aB  = (const uint4*)(A  + (size_t)mRow * K + kA);
  const uint4* b0B = (const uint4*)(BT + (size_t)(n0      + nCol) * K + kB);
  const uint4* b1B = (const uint4*)(BT + (size_t)(n0 + 16 + nCol) * K + kB);
  const uint4* b2B = (const uint4*)(BT + (size_t)(n0 + 32 + nCol) * K + kB);
  const uint4* b3B = (const uint4*)(BT + (size_t)(n0 + 48 + nCol) * K + kB);

  v8f acc0 = {}, acc1 = {}, acc2 = {}, acc3 = {};
  for (int k0 = 0; k0 < K; k0 += 32) {
    const int qi = k0 >> 3;                       // 32 bf16 = 4 uint4
    union { v16bf v; uint4 u[2]; } a, b0, b1, b2, b3;
    a.u[0]  = aB[qi];     a.u[1]  = aB[qi + 1];   // second run is K+8
    b0.u[0] = b0B[qi];    b0.u[1] = b0B[qi + 1];
    b1.u[0] = b1B[qi];    b1.u[1] = b1B[qi + 1];
    b2.u[0] = b2B[qi];    b2.u[1] = b2B[qi + 1];
    b3.u[0] = b3B[qi];    b3.u[1] = b3B[qi + 1];
    acc0 = __builtin_amdgcn_wmma_f32_16x16x32_bf16(false, a.v, false, b0.v, (short)0, acc0, false, false);
    acc1 = __builtin_amdgcn_wmma_f32_16x16x32_bf16(false, a.v, false, b1.v, (short)0, acc1, false, false);
    acc2 = __builtin_amdgcn_wmma_f32_16x16x32_bf16(false, a.v, false, b2.v, (short)0, acc2, false, false);
    acc3 = __builtin_amdgcn_wmma_f32_16x16x32_bf16(false, a.v, false, b3.v, (short)0, acc3, false, false);
  }

  // C/D layout: lanes 0-15: N=lane, M=r; lanes 16-31: N=lane-16, M=r+8
  const int mOff = (lane >> 4) << 3;
  union { v8f v; float f[8]; } r0, r1, r2, r3;
  r0.v = acc0; r1.v = acc1; r2.v = acc2; r3.v = acc3;
  const float bv0 = bias[n0 + nCol],      bv1 = bias[n0 + 16 + nCol];
  const float bv2 = bias[n0 + 32 + nCol], bv3 = bias[n0 + 48 + nCol];
  for (int r = 0; r < 8; ++r) {
    const size_t row = (size_t)(m0 + mOff + r) * N;
    C[row + n0      + nCol] = r0.f[r] + bv0;
    C[row + n0 + 16 + nCol] = r1.f[r] + bv1;
    C[row + n0 + 32 + nCol] = r2.f[r] + bv2;
    C[row + n0 + 48 + nCol] = r3.f[r] + bv3;
  }
}

// ---------------- persistent GRU scan: 16 blocks x 256 threads ----------------
__global__ void __launch_bounds__(256, 1)
k_gru_scan(const unsigned short* __restrict__ WhT,   // [3D, D] bf16 (r,z,n transposed)
           const float* __restrict__ G,              // [T, 3D] = x@W + b (all gates)
           const float* __restrict__ bhn,
           float* __restrict__ ys,                   // [T, D]
           float* __restrict__ hA, float* __restrict__ hB,
           int* __restrict__ bar) {
  __shared__ float hs[D_DIM];
  const int lane = threadIdx.x & 31;
  const int wvG  = blockIdx.x * (blockDim.x >> 5) + (threadIdx.x >> 5); // 0..127
  const int nWaves = gridDim.x * (blockDim.x >> 5);
  const int jPer = D_DIM / nWaves;                                      // 8
  const int jBase = wvG * jPer;

  for (int t = 0; t < T_DIM; ++t) {
    const float* hcur = (t & 1) ? hB : hA;
    float*       hnxt = (t & 1) ? hA : hB;
    for (int i = threadIdx.x; i < D_DIM; i += blockDim.x) hs[i] = hcur[i];
    __syncthreads();

    const float* Gt = G + (size_t)t * NG;
    if (t + 1 < T_DIM)
      __builtin_prefetch(G + (size_t)(t + 1) * NG + (threadIdx.x << 2), 0, 0);

    for (int jj = 0; jj < jPer; ++jj) {
      const int j = jBase + jj;
      const unsigned int* wr = (const unsigned int*)(WhT + (size_t)j * D_DIM);
      const unsigned int* wz = (const unsigned int*)(WhT + (size_t)(D_DIM + j) * D_DIM);
      const unsigned int* wn = (const unsigned int*)(WhT + (size_t)(2 * D_DIM + j) * D_DIM);
      float ar = 0.f, az = 0.f, an = 0.f;
      for (int i2 = lane; i2 < D_DIM / 2; i2 += 32) {   // packed pairs, 128B/wave loads
        const unsigned int ur = wr[i2], uz = wz[i2], un = wn[i2];
        const float h0 = hs[2 * i2], h1 = hs[2 * i2 + 1];
        ar = fmaf(bf2f((unsigned short)ur), h0, fmaf(bf2f((unsigned short)(ur >> 16)), h1, ar));
        az = fmaf(bf2f((unsigned short)uz), h0, fmaf(bf2f((unsigned short)(uz >> 16)), h1, az));
        an = fmaf(bf2f((unsigned short)un), h0, fmaf(bf2f((unsigned short)(un >> 16)), h1, an));
      }
      for (int o = 16; o; o >>= 1) {
        ar += __shfl_xor(ar, o, 32);
        az += __shfl_xor(az, o, 32);
        an += __shfl_xor(an, o, 32);
      }
      if (lane == 0) {
        const float r = 1.f / (1.f + __expf(-(Gt[j] + ar)));
        const float z = 1.f / (1.f + __expf(-(Gt[D_DIM + j] + az)));
        const float n = tanhf(Gt[2 * D_DIM + j] + r * (an + bhn[j]));
        const float hnew = (1.f - z) * n + z * hs[j];
        hnxt[j] = hnew;
        ys[(size_t)t * D_DIM + j] = hnew;
      }
    }

    // grid-wide monotonic barrier across 16 resident blocks (L2 atomics)
    __threadfence();
    __syncthreads();
    if (threadIdx.x == 0) {
      __hip_atomic_fetch_add(bar, 1, __ATOMIC_RELEASE, __HIP_MEMORY_SCOPE_AGENT);
      const int target = (t + 1) * (int)gridDim.x;
      while (__hip_atomic_load(bar, __ATOMIC_ACQUIRE, __HIP_MEMORY_SCOPE_AGENT) < target) {
        __builtin_amdgcn_s_sleep(1);
      }
    }
    __syncthreads();
  }
}

// ---------------- residual + layernorm ----------------
__global__ void __launch_bounds__(256)
k_ln(const float* __restrict__ xs, const float* __restrict__ lin,
     const float* __restrict__ scale, const float* __restrict__ bias,
     float* __restrict__ out) {
  const int t = blockIdx.x, tid = threadIdx.x;
  float v[4];
  float s = 0.f, s2 = 0.f;
  for (int c = 0; c < 4; ++c) {
    const int j = tid + c * 256;
    const float x = xs[(size_t)t * D_DIM + j] + lin[(size_t)t * D_DIM + j];
    v[c] = x; s += x; s2 += x * x;
  }
  for (int o = 16; o; o >>= 1) { s += __shfl_xor(s, o, 32); s2 += __shfl_xor(s2, o, 32); }
  __shared__ float w1[8], w2[8];
  __shared__ float mu_s, inv_s;
  if ((tid & 31) == 0) { w1[tid >> 5] = s; w2[tid >> 5] = s2; }
  __syncthreads();
  if (tid == 0) {
    float S = 0.f, S2 = 0.f;
    for (int i = 0; i < 8; ++i) { S += w1[i]; S2 += w2[i]; }
    const float mu = S * (1.f / D_DIM);
    mu_s = mu;
    inv_s = rsqrtf(S2 * (1.f / D_DIM) - mu * mu + 1e-6f);
  }
  __syncthreads();
  const float mu = mu_s, inv = inv_s;
  for (int c = 0; c < 4; ++c) {
    const int j = tid + c * 256;
    out[(size_t)t * D_DIM + j] = (v[c] - mu) * inv * scale[j] + bias[j];
  }
}

// ---------------- host launcher ----------------
extern "C" void kernel_launch(void* const* d_in, const int* in_sizes, int n_in,
                              void* d_out, int out_size, void* d_ws, size_t ws_size,
                              hipStream_t stream) {
  const float* xs         = (const float*)d_in[0];
  const float* init_state = (const float*)d_in[1];
  const float* Wir  = (const float*)d_in[2];
  const float* Wiz  = (const float*)d_in[3];
  const float* Win  = (const float*)d_in[4];
  const float* bir  = (const float*)d_in[5];
  const float* biz  = (const float*)d_in[6];
  const float* bin_ = (const float*)d_in[7];
  const float* Whr  = (const float*)d_in[8];
  const float* Whz  = (const float*)d_in[9];
  const float* Whn  = (const float*)d_in[10];
  const float* bhn  = (const float*)d_in[11];
  const float* Wl   = (const float*)d_in[12];
  const float* bl   = (const float*)d_in[13];
  const float* ln_scale = (const float*)d_in[14];
  const float* ln_bias  = (const float*)d_in[15];
  float* out = (float*)d_out;

  char* ws = (char*)d_ws;
  size_t off = 0;
  auto alloc = [&](size_t bytes) -> char* {
    char* p = ws + off; off += (bytes + 255) & ~(size_t)255; return p;
  };
  unsigned short* Xbf = (unsigned short*)alloc((size_t)T_DIM * D_DIM * 2);
  unsigned short* WgT = (unsigned short*)alloc((size_t)NG * D_DIM * 2);
  unsigned short* WhT = (unsigned short*)alloc((size_t)NG * D_DIM * 2);
  unsigned short* WlT = (unsigned short*)alloc((size_t)D_DIM * D_DIM * 2);
  float* G     = (float*)alloc((size_t)T_DIM * NG * 4);
  float* ysb   = (float*)alloc((size_t)T_DIM * D_DIM * 4);
  float* biasg = (float*)alloc((size_t)NG * 4);
  float* hA    = (float*)alloc((size_t)D_DIM * 4);
  float* hB    = (float*)alloc((size_t)D_DIM * 4);
  int*   bar   = (int*)alloc(256);
  unsigned short* Ybf = Xbf;   // reuse: Xbf dead after phase-1 GEMM
  float*          Lin = G;     // reuse: G dead after the scan

  const int n_td = T_DIM * D_DIM;
  k_init<<<4, 256, 0, stream>>>(init_state, bir, biz, bin_, hA, biasg, bar);
  k_f32_to_bf16<<<(n_td + 255) / 256, 256, 0, stream>>>(xs, Xbf, n_td);
  dim3 tg3(32, 32, 3), tg1(32, 32, 1);
  k_transpose_bf16<<<tg3, 1024, 0, stream>>>(Wir, Wiz, Win, WgT);
  k_transpose_bf16<<<tg3, 1024, 0, stream>>>(Whr, Whz, Whn, WhT);
  k_transpose_bf16<<<tg1, 1024, 0, stream>>>(Wl, Wl, Wl, WlT);

  // G = X @ [Wir|Wiz|Win] + biasg   (M=4096, N=3072, K=1024)
  const int waves1 = (T_DIM / 16) * (NG / 64);
  k_wmma_gemm<<<waves1 / 8, 256, 0, stream>>>(Xbf, WgT, biasg, G, T_DIM, NG, D_DIM);

  // sequential scan (persistent, 16 blocks, per-step L2 barrier)
  k_gru_scan<<<16, 256, 0, stream>>>(WhT, G, bhn, ysb, hA, hB, bar);

  // Lin = relu(ys) @ Wl + bl       (M=4096, N=1024, K=1024)
  k_relu_to_bf16<<<(n_td + 255) / 256, 256, 0, stream>>>(ysb, Ybf, n_td);
  const int waves2 = (T_DIM / 16) * (D_DIM / 64);
  k_wmma_gemm<<<waves2 / 8, 256, 0, stream>>>(Ybf, WlT, bl, Lin, T_DIM, D_DIM, D_DIM);

  // out = LN(xs + Lin) * scale + bias
  k_ln<<<T_DIM, 256, 0, stream>>>(xs, Lin, ln_scale, ln_bias, out);
}
